// DCNV4_Yolo11_18640158064725
// MI455X (gfx1250) — matrix operand order repaired and checked by reading, not compiled, not run on hardware
//
#include <hip/hip_runtime.h>
#include <hip/hip_bf16.h>
#include <math.h>

// ---------------- problem constants ----------------
#define NB      2
#define C_IN    256
#define CC      256
#define HH      80
#define WW      80
#define HW      (HH*WW)          // 6400
#define NHW     (NB*HW)          // 12800
#define GG      16
#define KS      3
#define KK      (KS*KS)          // 9
#define CG      (CC/GG)          // 16
#define OMW     (GG*KK*3)        // 432
#define OMW_P   512              // padded column count for guard-free WMMA
#define EPS_F   1e-05f

typedef _Float16 half_t;
typedef __attribute__((ext_vector_type(16))) _Float16 v16h;
typedef __attribute__((ext_vector_type(8)))  _Float16 v8h;
typedef __attribute__((ext_vector_type(8)))  float    v8f;

// ---------------- pack kernels ----------------

// x (N,C_IN,H,W) f32 -> xp [N*HW][C_IN] f16
__global__ void pack_x_kernel(const float* __restrict__ x, half_t* __restrict__ xp) {
    int idx = blockIdx.x * blockDim.x + threadIdx.x;   // over N*C_IN*HW
    if (idx >= NB * C_IN * HW) return;
    int hw = idx % HW;
    int c  = (idx / HW) % C_IN;
    int n  = idx / (HW * C_IN);
    float v = x[(size_t)(n * C_IN + c) * HW + hw];
    xp[(size_t)(n * HW + hw) * C_IN + c] = (half_t)v;
}

// W [K][Ncols] f32 -> WT [Npad][K] f16, zero fill rows nc >= Ncols
__global__ void pack_w_kernel(const float* __restrict__ w, half_t* __restrict__ wt,
                              int K, int Ncols, int Npad) {
    int idx = blockIdx.x * blockDim.x + threadIdx.x;   // over K*Npad
    if (idx >= K * Npad) return;
    int nc = idx % Npad;
    int k  = idx / Npad;
    float v = (nc < Ncols) ? w[(size_t)k * Ncols + nc] : 0.0f;
    wt[(size_t)nc * K + k] = (half_t)v;
}

// ---------------- WMMA 64x32 register-blocked strip core ----------------
// A: [M][K] f16 row-major.  BT: [Npad][K] f16 (B transposed).
// Wave computes 64 rows x 32 cols = 4x2 tiles of 16x16. No branches -> EXEC all-ones.
// D fragment: lane l holds N = 16*j + (l&15), M rows 16*i + (l>>4)*8 + r.
__device__ __forceinline__ void wmma_strip_64x32(const half_t* __restrict__ A,
                                                 const half_t* __restrict__ BT,
                                                 int K, int rowBase, int colStrip,
                                                 v8f acc[4][2]) {
    const int lane = threadIdx.x & 31;
    const int lid  = lane & 15;
    const int hs   = lane >> 4;
    const half_t* arow[4];
    const half_t* bcol[2];
#pragma unroll
    for (int i = 0; i < 4; ++i)
        arow[i] = A + (size_t)(rowBase + i * 16 + lid) * K;
#pragma unroll
    for (int j = 0; j < 2; ++j)
        bcol[j] = BT + (size_t)(colStrip + j * 16 + lid) * K;

    for (int kt = 0; kt < K; kt += 32) {
        v16h a[4], b[2];
#pragma unroll
        for (int i = 0; i < 4; ++i) {
            a[i] = *(const v16h*)(arow[i] + kt + hs * 16);
        }
#pragma unroll
        for (int j = 0; j < 2; ++j)
            b[j] = *(const v16h*)(bcol[j] + kt + hs * 16);
#pragma unroll
        for (int j = 0; j < 2; ++j)
#pragma unroll
            for (int i = 0; i < 4; ++i)
                acc[i][j] = __builtin_amdgcn_wmma_f32_16x16x32_f16(
                    false, a[i], false, b[j], (short)0, acc[i][j], false, false);
    }
}

__device__ __forceinline__ float silu_f(float v) {
    return v / (1.0f + __expf(-v));
}

// ---------------- GEMM1: h = xp @ w1, BN1 + SiLU, -> xf f16 ----------------
__global__ void gemm_bn_silu_kernel(const half_t* __restrict__ A, const half_t* __restrict__ BT,
                                    const float* __restrict__ g, const float* __restrict__ b,
                                    const float* __restrict__ m, const float* __restrict__ v,
                                    half_t* __restrict__ out) {
    const int wave     = threadIdx.x >> 5;
    const int rowBase  = blockIdx.x * 64;
    const int colStrip = (blockIdx.y * 4 + wave) * 32;
    v8f acc[4][2] = {};
    wmma_strip_64x32(A, BT, C_IN, rowBase, colStrip, acc);
    const int lane = threadIdx.x & 31;
    const int lid  = lane & 15;
    const int hs   = lane >> 4;
#pragma unroll
    for (int j = 0; j < 2; ++j) {
        const int c  = colStrip + j * 16 + lid;
        const float sc = rsqrtf(v[c] + EPS_F) * g[c];
        const float sh = b[c] - m[c] * sc;
#pragma unroll
        for (int i = 0; i < 4; ++i) {
            const int row0 = rowBase + i * 16 + hs * 8;
#pragma unroll
            for (int r = 0; r < 8; ++r) {
                float val = acc[i][j][r] * sc + sh;
                out[(size_t)(row0 + r) * CC + c] = (half_t)silu_f(val);
            }
        }
    }
}

// ---------------- GEMM2: value = A @ BT + bias -> f16 (stride Ncols) ----------------
__global__ void gemm_bias_f16_kernel(const half_t* __restrict__ A, const half_t* __restrict__ BT,
                                     const float* __restrict__ bias, half_t* __restrict__ out,
                                     int K, int Ncols) {
    const int wave     = threadIdx.x >> 5;
    const int rowBase  = blockIdx.x * 64;
    const int colStrip = (blockIdx.y * 4 + wave) * 32;
    v8f acc[4][2] = {};
    wmma_strip_64x32(A, BT, K, rowBase, colStrip, acc);
    const int lane = threadIdx.x & 31;
    const int lid  = lane & 15;
    const int hs   = lane >> 4;
#pragma unroll
    for (int j = 0; j < 2; ++j) {
        const int c   = colStrip + j * 16 + lid;
        const float bb = bias[c];
#pragma unroll
        for (int i = 0; i < 4; ++i) {
            const int row0 = rowBase + i * 16 + hs * 8;
#pragma unroll
            for (int r = 0; r < 8; ++r)
                out[(size_t)(row0 + r) * Ncols + c] = (half_t)(acc[i][j][r] + bb);
        }
    }
}

// ---------------- GEMM3: om = A @ BT + bias -> f32, padded cols ----------------
// bias has Nbias real entries; padded cols use bias 0 (results never read).
__global__ void gemm_bias_f32_kernel(const half_t* __restrict__ A, const half_t* __restrict__ BT,
                                     const float* __restrict__ bias, float* __restrict__ out,
                                     int K, int Npad, int Nbias) {
    const int wave     = threadIdx.x >> 5;
    const int rowBase  = blockIdx.x * 64;
    const int colStrip = (blockIdx.y * 4 + wave) * 32;
    v8f acc[4][2] = {};
    wmma_strip_64x32(A, BT, K, rowBase, colStrip, acc);
    const int lane = threadIdx.x & 31;
    const int lid  = lane & 15;
    const int hs   = lane >> 4;
#pragma unroll
    for (int j = 0; j < 2; ++j) {
        const int c   = colStrip + j * 16 + lid;
        const float bb = (c < Nbias) ? bias[c] : 0.0f;   // per-lane select, no branch
#pragma unroll
        for (int i = 0; i < 4; ++i) {
            const int row0 = rowBase + i * 16 + hs * 8;
#pragma unroll
            for (int r = 0; r < 8; ++r)
                out[(size_t)(row0 + r) * Npad + c] = acc[i][j][r] + bb;
        }
    }
}

// ---------------- deformable bilinear sampling ----------------
// value: [N*HW][CC] f16 (channel c = g*CG+cg); om: [N*HW][OMW_P] f32 (first 432 valid)
// acc -> [N*HW][CC] f16
__global__ void dcn_sample_kernel(const half_t* __restrict__ value,
                                  const float* __restrict__ om,
                                  half_t* __restrict__ accOut) {
    int idx = blockIdx.x * blockDim.x + threadIdx.x;   // over N*HW*G*CG
    if (idx >= NHW * GG * CG) return;
    const int cg  = idx & (CG - 1);
    const int gi  = (idx >> 4) & (GG - 1);
    const int pos = idx >> 8;               // n*HW + hw
    const int hw  = pos % HW;
    const int n   = pos / HW;
    const int yy  = hw / WW;
    const int xx  = hw % WW;

    const float* omg = om + (size_t)pos * OMW_P + gi * (3 * KK);
    const half_t* vbase = value + (size_t)n * HW * CC + gi * CG + cg;

    float acc = 0.0f;
#pragma unroll
    for (int k = 0; k < KK; ++k) {
        const float ox = omg[2 * k + 0];
        const float oy = omg[2 * k + 1];
        const float mk = omg[2 * KK + k];
        const float py = (float)yy - 1.0f + (float)(k / KS) + oy;
        const float px = (float)xx - 1.0f + (float)(k % KS) + ox;
        const float y0 = floorf(py), x0 = floorf(px);
        const float wy = py - y0,    wx = px - x0;
        const int y0i = (int)y0, x0i = (int)x0;
        const int y1i = y0i + 1, x1i = x0i + 1;
#pragma unroll
        for (int t = 0; t < 4; ++t) {
            const int yi = (t & 2) ? y1i : y0i;
            const int xi = (t & 1) ? x1i : x0i;
            const float wgt = ((t & 2) ? wy : (1.0f - wy)) * ((t & 1) ? wx : (1.0f - wx));
            const float valid = (yi >= 0 && yi < HH && xi >= 0 && xi < WW) ? 1.0f : 0.0f;
            const int yc = min(max(yi, 0), HH - 1);
            const int xc = min(max(xi, 0), WW - 1);
            const float vv = (float)vbase[(size_t)(yc * WW + xc) * CC];
            acc += mk * wgt * valid * vv;
        }
    }
    accOut[(size_t)pos * CC + gi * CG + cg] = (half_t)acc;
}

// ---------------- GEMM4: acc @ wo + bo, BN2 + SiLU, store NCHW f32 ----------------
__global__ void gemm_bn_silu_nchw_kernel(const half_t* __restrict__ A, const half_t* __restrict__ BT,
                                         const float* __restrict__ bias,
                                         const float* __restrict__ g, const float* __restrict__ b,
                                         const float* __restrict__ m, const float* __restrict__ v,
                                         float* __restrict__ out) {
    const int wave     = threadIdx.x >> 5;
    const int rowBase  = blockIdx.x * 64;     // global row = n*HW + hw ; HW % 64 == 0
    const int colStrip = (blockIdx.y * 4 + wave) * 32;
    v8f acc[4][2] = {};
    wmma_strip_64x32(A, BT, CC, rowBase, colStrip, acc);
    const int lane = threadIdx.x & 31;
    const int lid  = lane & 15;
    const int hs   = lane >> 4;
#pragma unroll
    for (int j = 0; j < 2; ++j) {
        const int c   = colStrip + j * 16 + lid;
        const float bb = bias[c];
        const float sc = rsqrtf(v[c] + EPS_F) * g[c];
        const float sh = b[c] - m[c] * sc;
#pragma unroll
        for (int i = 0; i < 4; ++i) {
            const int row0 = rowBase + i * 16 + hs * 8;   // stays within one batch image
            const int n    = row0 / HW;
            const int hw0  = row0 % HW;
            float* op = out + (size_t)(n * CC + c) * HW + hw0;  // 8 contiguous floats
#pragma unroll
            for (int r = 0; r < 8; ++r) {
                float val = (acc[i][j][r] + bb) * sc + sh;
                op[r] = silu_f(val);
            }
        }
    }
}

// ---------------- launcher ----------------
extern "C" void kernel_launch(void* const* d_in, const int* in_sizes, int n_in,
                              void* d_out, int out_size, void* d_ws, size_t ws_size,
                              hipStream_t stream) {
    const float* x     = (const float*)d_in[0];
    const float* w1    = (const float*)d_in[1];
    const float* bn1_g = (const float*)d_in[2];
    const float* bn1_b = (const float*)d_in[3];
    const float* bn1_m = (const float*)d_in[4];
    const float* bn1_v = (const float*)d_in[5];
    const float* wv    = (const float*)d_in[6];
    const float* bv    = (const float*)d_in[7];
    const float* wom   = (const float*)d_in[8];
    const float* bom   = (const float*)d_in[9];
    const float* wo    = (const float*)d_in[10];
    const float* bo    = (const float*)d_in[11];
    const float* bn2_g = (const float*)d_in[12];
    const float* bn2_b = (const float*)d_in[13];
    const float* bn2_m = (const float*)d_in[14];
    const float* bn2_v = (const float*)d_in[15];
    float* out = (float*)d_out;

    char* ws = (char*)d_ws;
    size_t off = 0;
    auto alloc = [&](size_t bytes) { char* p = ws + off; off = (off + bytes + 255) & ~(size_t)255; return p; };
    half_t* xp    = (half_t*)alloc((size_t)NHW * C_IN * 2);     // 6.55 MB
    half_t* xf    = (half_t*)alloc((size_t)NHW * CC   * 2);     // 6.55 MB
    half_t* value = (half_t*)alloc((size_t)NHW * CC   * 2);     // 6.55 MB
    half_t* accH  = (half_t*)alloc((size_t)NHW * CC   * 2);     // 6.55 MB
    half_t* w1T   = (half_t*)alloc((size_t)CC * C_IN  * 2);
    half_t* wvT   = (half_t*)alloc((size_t)CC * CC    * 2);
    half_t* woT   = (half_t*)alloc((size_t)CC * CC    * 2);
    half_t* womT  = (half_t*)alloc((size_t)OMW_P * CC * 2);     // padded to 512 rows
    float*  omBuf = (float*) alloc((size_t)NHW * OMW_P * 4);    // 26.2 MB
    (void)ws_size;

    // 1) pack inputs to f16 / transposed (zero-padded) layouts
    pack_x_kernel<<<(NB * C_IN * HW + 255) / 256, 256, 0, stream>>>(x, xp);
    pack_w_kernel<<<(C_IN * CC    + 255) / 256, 256, 0, stream>>>(w1,  w1T,  C_IN, CC,  CC);
    pack_w_kernel<<<(CC * CC      + 255) / 256, 256, 0, stream>>>(wv,  wvT,  CC,   CC,  CC);
    pack_w_kernel<<<(CC * OMW_P   + 255) / 256, 256, 0, stream>>>(wom, womT, CC,   OMW, OMW_P);
    pack_w_kernel<<<(CC * CC      + 255) / 256, 256, 0, stream>>>(wo,  woT,  CC,   CC,  CC);

    const int rowBlocks = NHW / 64;           // 200 (each block covers 64 rows)
    // Block: 4 waves x 32 cols = 128 cols. 256 cols -> grid.y=2; 512 -> grid.y=4.
    // 2) GEMM1 + BN1 + SiLU -> xf (f16)
    gemm_bn_silu_kernel<<<dim3(rowBlocks, 2), 128, 0, stream>>>(
        xp, w1T, bn1_g, bn1_b, bn1_m, bn1_v, xf);
    // 3) GEMM2: value = xf @ wv + bv (f16)
    gemm_bias_f16_kernel<<<dim3(rowBlocks, 2), 128, 0, stream>>>(
        xf, wvT, bv, value, CC, CC);
    // 4) GEMM3: om = xf @ wom + bom (f32), padded to 512 cols, guard-free
    gemm_bias_f32_kernel<<<dim3(rowBlocks, 4), 128, 0, stream>>>(
        xf, womT, bom, omBuf, CC, OMW_P, OMW);
    // 5) deformable sampling -> accH (f16)
    dcn_sample_kernel<<<(NHW * GG * CG + 255) / 256, 256, 0, stream>>>(value, omBuf, accH);
    // 6) GEMM4 + BN2 + SiLU, NCHW f32 output
    gemm_bn_silu_nchw_kernel<<<dim3(rowBlocks, 2), 128, 0, stream>>>(
        accH, woT, bo, bn2_g, bn2_b, bn2_m, bn2_v, out);
    (void)in_sizes; (void)n_in; (void)out_size;
}
